// BitLinear_68831145886302
// MI455X (gfx1250) — matrix operand order repaired and checked by reading, not compiled, not run on hardware
//
#include <hip/hip_runtime.h>
#include <hip/hip_bf16.h>
#include <stddef.h>

// ---------------------------------------------------------------------------
// BitLinear (ternary-quantized linear) for MI455X / gfx1250, wave32 + WMMA.
//   x      : [8, 2048, 4096] f32  -> treated as X[M=16384][K=4096]
//   weight : [16384, 4096]   f32  -> W[N][K]
//   scale  : [1] f32
//   out    : [8, 2048, 16384] f32 -> Out[M][N] = X @ Wq^T
// Pipeline: (1) deterministic two-pass |W| sum, (2) quantize W -> f16 Wq in
// workspace (exact ternary, fits in 192MB L2), (3) f16 WMMA GEMM, f32 accum.
// ---------------------------------------------------------------------------

typedef __attribute__((ext_vector_type(4)))  float    f32x4;
typedef __attribute__((ext_vector_type(8)))  float    f32x8;
typedef __attribute__((ext_vector_type(4)))  _Float16 f16x4;
typedef __attribute__((ext_vector_type(8)))  _Float16 f16x8;
typedef __attribute__((ext_vector_type(16))) _Float16 f16x16;

#define KDIM 4096
#define NDIM 16384
#define MDIM 16384

#define BM 128
#define BN 128
#define BK 32
#define BKP 40            // padded LDS row stride (halves): 80B rows, 16B aligned,
                          // 20-bank stride -> conflict-free b128 fragment loads
#define NK (KDIM / BK)    // 128 K-steps

#define RED_BLOCKS 2048

// ---------------------------------------------------------------------------
// Pass 1a: per-block partial sums of |W| (deterministic tree reduce).
// ---------------------------------------------------------------------------
__global__ __launch_bounds__(256)
void absum_partial(const float* __restrict__ W, float* __restrict__ partials) {
    __shared__ float red[256];
    const size_t total4 = (size_t)NDIM * KDIM / 4;
    const size_t stride = (size_t)gridDim.x * blockDim.x;
    float s = 0.0f;
    for (size_t i = (size_t)blockIdx.x * blockDim.x + threadIdx.x; i < total4; i += stride) {
        f32x4 v = ((const f32x4*)W)[i];
        s += __builtin_fabsf(v.x) + __builtin_fabsf(v.y)
           + __builtin_fabsf(v.z) + __builtin_fabsf(v.w);
    }
    red[threadIdx.x] = s;
    __syncthreads();
    for (int o = 128; o > 0; o >>= 1) {
        if ((int)threadIdx.x < o) red[threadIdx.x] += red[threadIdx.x + o];
        __syncthreads();
    }
    if (threadIdx.x == 0) partials[blockIdx.x] = red[0];
}

// ---------------------------------------------------------------------------
// Pass 1b: fold RED_BLOCKS partials into one scalar (fixed order -> replay-stable).
// ---------------------------------------------------------------------------
__global__ __launch_bounds__(256)
void absum_final(const float* __restrict__ partials, float* __restrict__ sum_out) {
    __shared__ float red[256];
    float s = 0.0f;
    for (int i = threadIdx.x; i < RED_BLOCKS; i += 256) s += partials[i];
    red[threadIdx.x] = s;
    __syncthreads();
    for (int o = 128; o > 0; o >>= 1) {
        if ((int)threadIdx.x < o) red[threadIdx.x] += red[threadIdx.x + o];
        __syncthreads();
    }
    if (threadIdx.x == 0) sum_out[0] = red[0];
}

// ---------------------------------------------------------------------------
// Pass 2: ternary-quantize W into f16 (exact: {-scale, 0, +scale}).
// ---------------------------------------------------------------------------
__global__ __launch_bounds__(256)
void quantize_w(const float* __restrict__ W, const float* __restrict__ sum_in,
                const float* __restrict__ scale_in, _Float16* __restrict__ Wq) {
    const float thr = 0.7f * sum_in[0] * (1.0f / ((float)NDIM * (float)KDIM));
    const float sc  = scale_in[0];
    const size_t total4 = (size_t)NDIM * KDIM / 4;
    const size_t stride = (size_t)gridDim.x * blockDim.x;
    for (size_t i = (size_t)blockIdx.x * blockDim.x + threadIdx.x; i < total4; i += stride) {
        f32x4 v = ((const f32x4*)W)[i];
        f16x4 h;
        #pragma unroll
        for (int j = 0; j < 4; ++j) {
            float w = v[j];
            float q = (__builtin_fabsf(w) > thr) ? (w > 0.0f ? sc : -sc) : 0.0f;
            h[j] = (_Float16)q;
        }
        ((f16x4*)Wq)[i] = h;
    }
}

// ---------------------------------------------------------------------------
// Pass 3: WMMA GEMM. Block 256 thr = 8 waves (4x2), wave tile 32x64,
// 2x4 = 8 v_wmma_f32_16x16x32_f16 per wave per K-step. Double-buffered LDS.
// ---------------------------------------------------------------------------
__global__ __launch_bounds__(256)
void bitlinear_gemm(const float* __restrict__ X, const _Float16* __restrict__ Wq,
                    float* __restrict__ Out) {
    __shared__ _Float16 sA[2][BM * BKP];   // x tile, f16, padded rows
    __shared__ _Float16 sB[2][BN * BKP];   // Wq tile, padded rows

    const int t    = threadIdx.x;
    const int lane = t & 31;
    const int wave = t >> 5;
    const int wm   = wave >> 1;            // 0..3 -> row group of 32
    const int wn   = wave & 1;             // 0..1 -> col group of 64
    const int blockM = blockIdx.y * BM;
    const int blockN = blockIdx.x * BN;

    // Global-load staging registers (A: 16 f32 / thread, B: 16 halves / thread)
    f32x4 aReg[4];
    f16x8 bReg[2];

    auto loadGlobal = [&](int kk) {
        #pragma unroll
        for (int i = 0; i < 4; ++i) {                // 1024 float4 per tile
            int lin = t + i * 256;
            int row = lin >> 3, c4 = lin & 7;        // 8 float4 per 32-wide row
            aReg[i] = *(const f32x4*)(X + (size_t)(blockM + row) * KDIM + kk + c4 * 4);
        }
        #pragma unroll
        for (int i = 0; i < 2; ++i) {                // 512 x 8-half chunks per tile
            int lin = t + i * 256;
            int row = lin >> 2, c8 = lin & 3;        // 4 chunks per 32-wide row
            bReg[i] = *(const f16x8*)(Wq + (size_t)(blockN + row) * KDIM + kk + c8 * 8);
        }
    };

    auto storeLds = [&](int s) {
        #pragma unroll
        for (int i = 0; i < 4; ++i) {
            int lin = t + i * 256;
            int row = lin >> 3, c4 = lin & 7;
            f16x4 h = __builtin_convertvector(aReg[i], f16x4);   // f32 -> f16 here
            *(f16x4*)&sA[s][row * BKP + c4 * 4] = h;             // 8B store, aligned
        }
        #pragma unroll
        for (int i = 0; i < 2; ++i) {
            int lin = t + i * 256;
            int row = lin >> 2, c8 = lin & 3;
            *(f16x8*)&sB[s][row * BKP + c8 * 8] = bReg[i];       // 16B store, aligned
        }
    };

    // Prologue: tile 0
    loadGlobal(0);
    storeLds(0);
    __syncthreads();

    f32x8 acc[2][4] = {};
    const int halfSel = (lane < 16) ? 0 : 1;
    const int kbA = halfSel * 8;     // ISA 16-bit A layout: lanes 16-31 hold K=8..15 / 24..31
    const int kbB = halfSel * 8;     // ISA 16-bit B layout: lanes 16-31 hold K=8..15 / 24..31
    const int l16 = lane & 15;

    int cur = 0;
    for (int kt = 0; kt < NK; ++kt) {
        const bool more = (kt + 1) < NK;
        if (more) loadGlobal((kt + 1) * BK);
        if (kt + 2 < NK) {           // pull K+2 strips toward L2 (global_prefetch_b8)
            const int kk2 = (kt + 2) * BK;
            __builtin_prefetch(X  + (size_t)(blockM + (t >> 1)) * KDIM + kk2, 0, 1);
            __builtin_prefetch(Wq + (size_t)(blockN + (t >> 1)) * KDIM + kk2, 0, 1);
        }

        // --- fragments from LDS per ISA 7.12.2 layouts ---
        f16x16 af[2];
        #pragma unroll
        for (int im = 0; im < 2; ++im) {
            int r = wm * 32 + im * 16 + l16;
            f16x8 lo = *(const f16x8*)&sA[cur][r * BKP + kbA];        // K = kb..kb+7
            f16x8 hi = *(const f16x8*)&sA[cur][r * BKP + kbA + 16];   // K = kb+16..kb+23
            af[im] = __builtin_shufflevector(lo, hi, 0,1,2,3,4,5,6,7,8,9,10,11,12,13,14,15);
        }
        f16x16 bf[4];
        #pragma unroll
        for (int in = 0; in < 4; ++in) {
            int c = wn * 64 + in * 16 + l16;
            f16x8 lo = *(const f16x8*)&sB[cur][c * BKP + kbB];        // K = kb..kb+7
            f16x8 hi = *(const f16x8*)&sB[cur][c * BKP + kbB + 16];   // K = kb+16..kb+23
            bf[in] = __builtin_shufflevector(lo, hi, 0,1,2,3,4,5,6,7,8,9,10,11,12,13,14,15);
        }

        #pragma unroll
        for (int im = 0; im < 2; ++im)
            #pragma unroll
            for (int in = 0; in < 4; ++in)
                acc[im][in] = __builtin_amdgcn_wmma_f32_16x16x32_f16(
                    /*neg_a=*/false, af[im], /*neg_b=*/false, bf[in],
                    /*c_mod=*/(short)0, acc[im][in],
                    /*reuse_a=*/false, /*reuse_b=*/false);

        if (more) storeLds(cur ^ 1);   // write other stage; nobody reads it yet
        __syncthreads();               // single barrier per K-step
        cur ^= 1;
    }

    // Epilogue: ISA C/D layout — VGPR r: lanes 0-15 -> M=r, lanes 16-31 -> M=8+r.
    const int rowSel = halfSel * 8;
    #pragma unroll
    for (int im = 0; im < 2; ++im) {
        const int r0 = blockM + wm * 32 + im * 16 + rowSel;
        #pragma unroll
        for (int in = 0; in < 4; ++in) {
            const int c = blockN + wn * 64 + in * 16 + l16;
            #pragma unroll
            for (int r = 0; r < 8; ++r)
                Out[(size_t)(r0 + r) * NDIM + c] = acc[im][in][r];
        }
    }
}

// ---------------------------------------------------------------------------
// Launcher. Workspace layout:
//   [0 .. 8KB)          : 2048 partial sums (f32)
//   [8KB .. 8KB+4)      : |W| sum scalar
//   [16KB ...)          : Wq f16, 16384*4096*2 = 128 MB
// ---------------------------------------------------------------------------
extern "C" void kernel_launch(void* const* d_in, const int* in_sizes, int n_in,
                              void* d_out, int out_size, void* d_ws, size_t ws_size,
                              hipStream_t stream) {
    (void)in_sizes; (void)n_in; (void)out_size; (void)ws_size;
    const float* X     = (const float*)d_in[0];
    const float* W     = (const float*)d_in[1];
    const float* scale = (const float*)d_in[2];
    float*       Out   = (float*)d_out;

    char* ws = (char*)d_ws;
    float*    partials = (float*)ws;
    float*    sum      = (float*)(ws + RED_BLOCKS * sizeof(float));
    _Float16* Wq       = (_Float16*)(ws + 16384);

    absum_partial<<<RED_BLOCKS, 256, 0, stream>>>(W, partials);
    absum_final  <<<1,          256, 0, stream>>>(partials, sum);
    quantize_w   <<<8192,       256, 0, stream>>>(W, sum, scale, Wq);

    dim3 grid(NDIM / BN, MDIM / BM);   // 128 x 128 blocks
    bitlinear_gemm<<<grid, 256, 0, stream>>>(X, Wq, Out);
}
